// LWAPlusPlus_42133629174171
// MI455X (gfx1250) — matrix operand order repaired and checked by reading, not compiled, not run on hardware
//
#include <hip/hip_runtime.h>
#include <hip/hip_bf16.h>

typedef __attribute__((ext_vector_type(16))) _Float16 v16h;
typedef __attribute__((ext_vector_type(8)))  float    v8f;

// ---------------------------------------------------------------------------
// Sizes (fixed per reference): B=4, C=64, H=W=256, K=7, NOR=8, GH=64
// ---------------------------------------------------------------------------
#define HB 4
#define HC 64
#define HH 256
#define HW 256
#define HHH 128           // half res
#define HHW 128
#define NOUT1 (16777216)  // 4*64*256*256

// Generic->LDS addrspace cast: yields the wave-relative LDS byte offset that
// GLOBAL_LOAD_ASYNC_TO_LDS expects in its VDST register.
__device__ __forceinline__ uint32_t lds_off(const void* p) {
  return (uint32_t)(uintptr_t)(__attribute__((address_space(3))) const void*)p;
}

// ---------------------------------------------------------------------------
// Kernel 0: repack gw1/gw2 (fp32, OIHW) into WMMA A-matrix per-lane f16 order
//   A 16x32 layout: lane l -> M = l&15 ; half i -> K = (l<16?0:16) + i
//   apack1[t][ck][ctile][lane][i]  : 9*8*4*32*16  (3x3 conv, Cin=256, Cout=64)
//   apack2[ck][ctile][lane][i]     : 2*16*32*16   (1x1 conv, Cin=64,  Cout=256)
// ---------------------------------------------------------------------------
__global__ void pack_kernel(const float* __restrict__ gw1,
                            const float* __restrict__ gw2,
                            _Float16* __restrict__ apack1,
                            _Float16* __restrict__ apack2) {
  int tid = blockIdx.x * 256 + threadIdx.x;
  const int N1 = 9 * 8 * 4 * 32 * 16;   // 147456
  const int N2 = 2 * 16 * 32 * 16;      // 16384
  if (tid < N1) {
    int i     = tid & 15;
    int lane  = (tid >> 4) & 31;
    int ctile = (tid >> 9) & 3;
    int ck    = (tid >> 11) & 7;
    int t     = tid >> 14;              // 0..8
    int M  = lane & 15;
    int kb = (lane < 16) ? 0 : 16;
    int K  = kb + i;
    int o  = ctile * 16 + M;
    int ic = ck * 32 + K;
    int dy = t / 3, dx = t - dy * 3;
    apack1[tid] = (_Float16)gw1[(((size_t)o * 256 + ic) * 3 + dy) * 3 + dx];
  } else if (tid < N1 + N2) {
    int u     = tid - N1;
    int i     = u & 15;
    int lane  = (u >> 4) & 31;
    int ctile = (u >> 9) & 15;
    int ck    = u >> 13;                // 0..1
    int M  = lane & 15;
    int kb = (lane < 16) ? 0 : 16;
    int K  = kb + i;
    int o  = ctile * 16 + M;
    int ic = ck * 32 + K;
    apack2[u] = (_Float16)gw2[(size_t)o * 64 + ic];
  }
}

// ---------------------------------------------------------------------------
// Kernel 1: depthwise 7x7 line conv (SAME, zero pad), max over 8 orientations.
// ---------------------------------------------------------------------------
__global__ __launch_bounds__(256)
void lineconv_kernel(const float* __restrict__ x, const float* __restrict__ lw,
                     float* __restrict__ line_r) {
  __shared__ float xt[22 * 22];
  __shared__ float wv[8 * 49];
  int blk = blockIdx.x;
  int tX = blk & 15;
  int tY = (blk >> 4) & 15;
  int c  = (blk >> 8) & 63;
  int b  = blk >> 14;
  int tid = threadIdx.x;
  const float* xc = x + ((size_t)(b * HC + c) << 16);
  int y0 = tY * 16, x0 = tX * 16;
  for (int it = tid; it < 484; it += 256) {
    int iy = it / 22, ix = it - iy * 22;
    int gy = y0 + iy - 3, gx = x0 + ix - 3;
    float v = 0.f;
    if (gy >= 0 && gy < HH && gx >= 0 && gx < HW) v = xc[gy * HW + gx];
    xt[it] = v;
  }
  for (int it = tid; it < 392; it += 256) {
    int o = it / 49, tap = it - o * 49;
    wv[it] = lw[((size_t)o * HC + c) * 49 + tap];
  }
  __syncthreads();
  int ty = tid >> 4, tx = tid & 15;
  float acc[8] = {0.f, 0.f, 0.f, 0.f, 0.f, 0.f, 0.f, 0.f};
  for (int ky = 0; ky < 7; ++ky) {
    for (int kx = 0; kx < 7; ++kx) {
      float v = xt[(ty + ky) * 22 + (tx + kx)];
      int tap = ky * 7 + kx;
#pragma unroll
      for (int o = 0; o < 8; ++o) acc[o] = fmaf(wv[o * 49 + tap], v, acc[o]);
    }
  }
  float m = acc[0];
#pragma unroll
  for (int o = 1; o < 8; ++o) m = fmaxf(m, acc[o]);
  line_r[((size_t)(b * HC + c) << 16) + (y0 + ty) * HW + (x0 + tx)] = m;
}

// ---------------------------------------------------------------------------
// Kernel 2: build gate_in (NHWC f16, 256 ch @ 128x128):
//   ch [0,64)=lh, [64,128)=hl, [128,192)=hh, [192,256)=2x2-avg(line_r)
// ---------------------------------------------------------------------------
__global__ void gatein_kernel(const float* __restrict__ x,
                              const float* __restrict__ line_r,
                              _Float16* __restrict__ gi) {
  int tid = blockIdx.x * 256 + threadIdx.x;
  if (tid >= HB * HC * HHH * HHW) return;
  int xq = tid & 127;
  int yq = (tid >> 7) & 127;
  int c  = (tid >> 14) & 63;
  int b  = tid >> 20;
  size_t base = ((size_t)(b * HC + c) << 16) + (size_t)(yq * 2) * HW + xq * 2;
  float x00 = x[base],       x01 = x[base + 1];
  float x10 = x[base + HW],  x11 = x[base + HW + 1];
  float lh = 0.5f * (x00 + x01 - x10 - x11);
  float hl = 0.5f * (x00 - x01 + x10 - x11);
  float hh = 0.5f * (x00 - x01 - x10 + x11);
  float l00 = line_r[base],      l01 = line_r[base + 1];
  float l10 = line_r[base + HW], l11 = line_r[base + HW + 1];
  float ld = 0.25f * (l00 + l01 + l10 + l11);
  size_t p = (((size_t)(b * HHH + yq) * HHW) + xq) * 256;
  gi[p + c]       = (_Float16)lh;
  gi[p + 64 + c]  = (_Float16)hl;
  gi[p + 128 + c] = (_Float16)hh;
  gi[p + 192 + c] = (_Float16)ld;
}

// ---------------------------------------------------------------------------
// Kernel 3: implicit-GEMM 3x3 conv 256->64 + gelu, via v_wmma_f32_16x16x32_f16
//   Double-buffered LDS staging with GLOBAL_LOAD_ASYNC_TO_LDS_B128 (ASYNCcnt):
//   the async DMA for channel-chunk ck+1 overlaps the 9 WMMAs of chunk ck.
// ---------------------------------------------------------------------------
__global__ __launch_bounds__(128)
void conv1_kernel(const _Float16* __restrict__ gi,
                  const _Float16* __restrict__ apack,
                  const float* __restrict__ gb1,
                  _Float16* __restrict__ hmid) {
  __shared__ alignas(32) _Float16 tile[2][3 * 18 * 32];
  int blk  = blockIdx.x;          // b*1024 + tileId
  int b    = blk >> 10;
  int t128 = blk & 1023;
  int y0 = t128 >> 3;
  int x0 = (t128 & 7) << 4;
  int tid  = threadIdx.x;
  int wave = tid >> 5;
  int lane = tid & 31;

  v8f acc = {0.f, 0.f, 0.f, 0.f, 0.f, 0.f, 0.f, 0.f};
  int n   = lane & 15;            // spatial pixel within tile (B-matrix N)
  int kbB = (lane < 16) ? 0 : 16; // B-matrix K base within 32-chunk

  // Issue async memory->LDS fill of one 3x18x32-channel NHWC tile.
  auto issue_fill = [&](int ck, int bufsel) {
    for (int item = tid; item < 216; item += 128) {
      int p  = item >> 2;
      int q  = item & 3;
      int yy = p / 18;
      int xx = p - yy * 18;
      int gy = y0 + yy - 1;
      int gx = x0 + xx - 1;
      _Float16* dst = &tile[bufsel][p * 32 + q * 8];
      if (gy >= 0 && gy < HHH && gx >= 0 && gx < HHW) {
        const _Float16* src =
            gi + (((size_t)(b * HHH + gy) * HHW + gx) * 256 + ck * 32 + q * 8);
        uint32_t ldsAddr = lds_off(dst);
        uint64_t gAddr   = (uint64_t)(uintptr_t)src;
        asm volatile("global_load_async_to_lds_b128 %0, %1, off"
                     :: "v"(ldsAddr), "v"(gAddr) : "memory");
      } else {
        uint4 z = {0u, 0u, 0u, 0u};
        *reinterpret_cast<uint4*>(dst) = z;   // zero halo via DS store
      }
    }
  };

  issue_fill(0, 0);
  asm volatile("s_wait_asynccnt 0x0" ::: "memory");
  __syncthreads();

  for (int ck = 0; ck < 8; ++ck) {
    if (ck < 7) issue_fill(ck + 1, (ck + 1) & 1);  // overlap DMA with WMMA
    const _Float16* buf = tile[ck & 1];
#pragma unroll
    for (int t = 0; t < 9; ++t) {
      int dy = t / 3, dx = t - dy * 3;
      v16h A = *reinterpret_cast<const v16h*>(
          apack + (((size_t)(t * 8 + ck) * 4 + wave) * 32 + lane) * 16);
      v16h B = *reinterpret_cast<const v16h*>(&buf[(dy * 18 + n + dx) * 32 + kbB]);
      acc = __builtin_amdgcn_wmma_f32_16x16x32_f16(false, A, false, B,
                                                   (short)0, acc, false, false);
    }
    asm volatile("s_wait_asynccnt 0x0" ::: "memory");
    __syncthreads();
  }

  // D: lane -> pixel x0+(lane&15); vgpr v -> outch wave*16 + v + 8*(lane>=16)
  int px     = x0 + n;
  int chBase = wave * 16 + ((lane < 16) ? 0 : 8);
  size_t pixIdx = (size_t)(b * HHH + y0) * HHW + px;
  alignas(16) _Float16 outv[8];
#pragma unroll
  for (int v = 0; v < 8; ++v) {
    float z = acc[v] + gb1[chBase + v];
    outv[v] = (_Float16)(0.5f * z * (1.0f + erff(z * 0.70710678118654752f)));
  }
  *reinterpret_cast<uint4*>(hmid + pixIdx * 64 + chBase) =
      *reinterpret_cast<uint4*>(outv);
}

// ---------------------------------------------------------------------------
// Kernel 4: 1x1 conv 64->256 + sigmoid, via v_wmma_f32_16x16x32_f16.
//   Writes g straight into d_out (fp32, NCHW).
// ---------------------------------------------------------------------------
__global__ __launch_bounds__(128)
void conv2_kernel(const _Float16* __restrict__ hmid,
                  const _Float16* __restrict__ apack2,
                  const float* __restrict__ gb2,
                  float* __restrict__ g) {
  int blk  = blockIdx.x;
  int b    = blk >> 10;
  int t128 = blk & 1023;
  int y0 = t128 >> 3;
  int x0 = (t128 & 7) << 4;
  int tid  = threadIdx.x;
  int wave = tid >> 5;
  int lane = tid & 31;
  int n    = lane & 15;
  int kbB  = (lane < 16) ? 0 : 16;
  int px   = x0 + n;
  size_t pixIdx = (size_t)(b * HHH + y0) * HHW + px;
  v16h B0 = *reinterpret_cast<const v16h*>(hmid + pixIdx * 64 + 0 * 32 + kbB);
  v16h B1 = *reinterpret_cast<const v16h*>(hmid + pixIdx * 64 + 1 * 32 + kbB);
#pragma unroll
  for (int rep = 0; rep < 4; ++rep) {
    int ctile = wave + rep * 4;   // 0..15 across the block
    v8f acc = {0.f, 0.f, 0.f, 0.f, 0.f, 0.f, 0.f, 0.f};
    v16h A0 = *reinterpret_cast<const v16h*>(
        apack2 + (((size_t)(0 * 16 + ctile) * 32 + lane) * 16));
    acc = __builtin_amdgcn_wmma_f32_16x16x32_f16(false, A0, false, B0,
                                                 (short)0, acc, false, false);
    v16h A1 = *reinterpret_cast<const v16h*>(
        apack2 + (((size_t)(1 * 16 + ctile) * 32 + lane) * 16));
    acc = __builtin_amdgcn_wmma_f32_16x16x32_f16(false, A1, false, B1,
                                                 (short)0, acc, false, false);
    int chBase = ctile * 16 + ((lane < 16) ? 0 : 8);
#pragma unroll
    for (int v = 0; v < 8; ++v) {
      int ch = chBase + v;
      float z = acc[v] + gb2[ch];
      float s = 1.0f / (1.0f + expf(-z));
      g[((size_t)(b * 256 + ch) << 14) + (y0 << 7) + px] = s;
    }
  }
}

// ---------------------------------------------------------------------------
// Kernel 5: fuse IDWT(gated hf) + bilinear-upsampled gline*line_r + blend.
// ---------------------------------------------------------------------------
__global__ void final_kernel(const float* __restrict__ x,
                             const float* __restrict__ line_r,
                             const float* __restrict__ g,
                             const float* __restrict__ scale,
                             float* __restrict__ out) {
  int tid = blockIdx.x * 256 + threadIdx.x;
  if (tid >= HB * HC * HHH * HHW) return;
  int xq = tid & 127;
  int yq = (tid >> 7) & 127;
  int c  = (tid >> 14) & 63;
  int b  = tid >> 20;
  size_t base = ((size_t)(b * HC + c) << 16) + (size_t)(yq * 2) * HW + xq * 2;
  float x00 = x[base],      x01 = x[base + 1];
  float x10 = x[base + HW], x11 = x[base + HW + 1];
  float ll = 0.5f * (x00 + x01 + x10 + x11);
  float lh = 0.5f * (x00 + x01 - x10 - x11);
  float hl = 0.5f * (x00 - x01 + x10 - x11);
  float hh = 0.5f * (x00 - x01 - x10 + x11);
  size_t ghalf = (size_t)yq * HHW + xq;
  size_t gbase = ((size_t)b * 256) << 14;
  float glh = g[gbase + ((size_t)(c) << 14) + ghalf];
  float ghl = g[gbase + ((size_t)(64 + c) << 14) + ghalf];
  float ghh = g[gbase + ((size_t)(128 + c) << 14) + ghalf];
  float lh2 = lh * (1.f + glh);
  float hl2 = hl * (1.f + ghl);
  float hh2 = hh * (1.f + ghh);
  float h00 = 0.5f * (ll + lh2 + hl2 + hh2);
  float h01 = 0.5f * (ll + lh2 - hl2 - hh2);
  float h10 = 0.5f * (ll - lh2 + hl2 - hh2);
  float h11 = 0.5f * (ll - lh2 - hl2 + hh2);
  // bilinear up (half-pixel centers, edge clamp)
  const float* gl = g + gbase + ((size_t)(192 + c) << 14);
  int ym = yq > 0 ? yq - 1 : 0, yp = yq < 127 ? yq + 1 : 127;
  int xm = xq > 0 ? xq - 1 : 0, xp = xq < 127 ? xq + 1 : 127;
  float gmm = gl[ym * 128 + xm], gm0 = gl[ym * 128 + xq], gmp = gl[ym * 128 + xp];
  float g0m = gl[yq * 128 + xm], gcc = gl[yq * 128 + xq], g0p = gl[yq * 128 + xp];
  float gpm = gl[yp * 128 + xm], gp0 = gl[yp * 128 + xq], gpp = gl[yp * 128 + xp];
  float u00 = 0.0625f * gmm + 0.1875f * gm0 + 0.1875f * g0m + 0.5625f * gcc;
  float u01 = 0.1875f * gm0 + 0.0625f * gmp + 0.5625f * gcc + 0.1875f * g0p;
  float u10 = 0.1875f * g0m + 0.5625f * gcc + 0.0625f * gpm + 0.1875f * gp0;
  float u11 = 0.5625f * gcc + 0.1875f * g0p + 0.1875f * gp0 + 0.0625f * gpp;
  float l00 = line_r[base],      l01 = line_r[base + 1];
  float l10 = line_r[base + HW], l11 = line_r[base + HW + 1];
  float s = scale[0];
  out[base]          = x00 + s * (0.5f * (h00 + x00 + u00 * l00) - x00);
  out[base + 1]      = x01 + s * (0.5f * (h01 + x01 + u01 * l01) - x01);
  out[base + HW]     = x10 + s * (0.5f * (h10 + x10 + u10 * l10) - x10);
  out[base + HW + 1] = x11 + s * (0.5f * (h11 + x11 + u11 * l11) - x11);
}

// ---------------------------------------------------------------------------
extern "C" void kernel_launch(void* const* d_in, const int* in_sizes, int n_in,
                              void* d_out, int out_size, void* d_ws, size_t ws_size,
                              hipStream_t stream) {
  (void)in_sizes; (void)n_in; (void)out_size; (void)ws_size;
  const float* x      = (const float*)d_in[0];
  const float* line_w = (const float*)d_in[1];
  const float* gw1    = (const float*)d_in[2];
  const float* gb1    = (const float*)d_in[3];
  const float* gw2    = (const float*)d_in[4];
  const float* gb2    = (const float*)d_in[5];
  const float* scale  = (const float*)d_in[6];
  float* out = (float*)d_out;
  float* g   = out + NOUT1;             // second tuple element, fp32 NCHW

  char* ws = (char*)d_ws;
  float*    line_r  = (float*)(ws);                       // 64 MB
  _Float16* gate_in = (_Float16*)(ws + 67108864);         // 32 MB  NHWC f16
  _Float16* hmid    = (_Float16*)(ws + 100663296);        //  8 MB  NHWC f16
  _Float16* apack1  = (_Float16*)(ws + 109051904);        // 288 KB
  _Float16* apack2  = (_Float16*)(ws + 109346816);        //  32 KB

  pack_kernel<<<640, 256, 0, stream>>>(gw1, gw2, apack1, apack2);
  lineconv_kernel<<<65536, 256, 0, stream>>>(x, line_w, line_r);
  gatein_kernel<<<16384, 256, 0, stream>>>(x, line_r, gate_in);
  conv1_kernel<<<4096, 128, 0, stream>>>(gate_in, apack1, gb1, hmid);
  conv2_kernel<<<4096, 128, 0, stream>>>(hmid, apack2, gb2, g);
  final_kernel<<<16384, 256, 0, stream>>>(x, line_r, g, scale, out);
}
